// MHA_17910013624928
// MI455X (gfx1250) — matrix operand order repaired and checked by reading, not compiled, not run on hardware
//
#include <hip/hip_runtime.h>

typedef __bf16 bf16;
typedef __attribute__((ext_vector_type(16))) __bf16 v16bf;
typedef __attribute__((ext_vector_type(8)))  __bf16 v8bf;
typedef __attribute__((ext_vector_type(2)))  __bf16 v2bf;
typedef __attribute__((ext_vector_type(8)))  float  v8f;

static constexpr int BATCH = 2;
static constexpr int SEQ   = 2048;
static constexpr int EMB   = 2048;
static constexpr int NH    = 32;   // query heads
static constexpr int NKV   = 8;    // kv heads
static constexpr int HD    = 64;   // head dim
static constexpr int NHT   = NH + 2 * NKV;          // 48 total qkv heads
static constexpr int QKVN  = NHT * HD;              // 3072
static constexpr float LOG_BASE_OVER_HALF_ROT = 0.5756462732485115f; // ln(10000)/16

// ---------------------------------------------------------------------------
// WMMA fragment loaders (bf16, 16x16x32). Per CDNA5 ISA 7.12.2:
//  A (16xK, row-major, ld in elems): lane = M(0..15) | kgroup(lane>>4);
//    elements 0..7  -> K = 8*kg + 0..7      (16 contiguous bytes)
//    elements 8..15 -> K = 16 + 8*kg + 0..7 (16 contiguous bytes)
//  B (Kx16): loaded from B^T row-major [N][K]; lane = N | kgroup;
//    elements 0..7  -> K = 8*kg + 0..7; elements 8..15 -> K = 16 + 8*kg + 0..7
// ---------------------------------------------------------------------------
__device__ __forceinline__ v16bf load_a_frag(const bf16* p, int ld) {
  const int lane = threadIdx.x & 31;
  const bf16* r = p + (size_t)(lane & 15) * ld + ((lane >> 4) << 3);
  v8bf lo = *(const v8bf*)(r);
  v8bf hi = *(const v8bf*)(r + 16);
  return __builtin_shufflevector(lo, hi, 0,1,2,3,4,5,6,7,8,9,10,11,12,13,14,15);
}

__device__ __forceinline__ v16bf load_b_frag(const bf16* p, int ld) {
  const int lane = threadIdx.x & 31;
  const bf16* r = p + (size_t)(lane & 15) * ld + ((lane >> 4) << 3);
  v8bf lo = *(const v8bf*)(r);
  v8bf hi = *(const v8bf*)(r + 16);
  return __builtin_shufflevector(lo, hi, 0,1,2,3,4,5,6,7,8,9,10,11,12,13,14,15);
}

__device__ __forceinline__ v8f wmma_bf16(v16bf a, v16bf b, v8f c) {
  return __builtin_amdgcn_wmma_f32_16x16x32_bf16(
      /*neg_a=*/false, a, /*neg_b=*/false, b,
      /*c_mod=*/(short)0, c, /*reuse_a=*/false, /*reuse_b=*/false);
}

// ---------------------------------------------------------------------------
// GEMM: C[M,N] = A[M,K](TA) * B[K,N](f32) + bias[N], bf16 WMMA compute.
// Block tile 128x128, 8 waves (4 along M x 2 along N), each wave 32x64.
// Software-pipelined: next K-slice global loads issue while WMMAs run.
// LDS stores are packed (b128 for A, b32 pairs for transposed B).
// ---------------------------------------------------------------------------
template <typename TA>
__global__ __launch_bounds__(256)
void gemm_bias_kernel(const TA* __restrict__ A, const float* __restrict__ Bm,
                      const float* __restrict__ bias, float* __restrict__ C,
                      int M, int N, int K) {
  __shared__ bf16 As[128 * 32];  // [m][k], ld 32
  __shared__ bf16 Bs[128 * 32];  // B^T tile: [n][k], ld 32

  const int bm  = blockIdx.x * 128;
  const int bn  = blockIdx.y * 128;
  const int tid = threadIdx.x;
  const int wave = tid >> 5;
  const int wm = (wave & 3) * 32;   // wave's M offset in tile
  const int wn = (wave >> 2) * 64;  // wave's N offset in tile

  // Staging assignment:
  //  A: thread -> row tid>>1, 16 contiguous k at (tid&1)*16
  //  B: thread -> k-row pair (tid&15)*2, 8 contiguous n at (tid>>4)*8
  const int arow = tid >> 1;
  const int acol = (tid & 1) << 4;
  const int bkk  = (tid & 15) << 1;
  const int bnn  = (tid >> 4) << 3;

  float4 aRegF[4];
  v8bf   aRegH[2];
  float4 bReg[4];

  auto load_stage = [&](int k0) {
    if constexpr (sizeof(TA) == 4) {
      const float4* g = (const float4*)(A + (size_t)(bm + arow) * K + k0 + acol);
      #pragma unroll
      for (int i = 0; i < 4; ++i) aRegF[i] = g[i];
    } else {
      const v8bf* g = (const v8bf*)(A + (size_t)(bm + arow) * K + k0 + acol);
      aRegH[0] = g[0];
      aRegH[1] = g[1];
    }
    const float* g0 = Bm + (size_t)(k0 + bkk) * N + bn + bnn;
    bReg[0] = ((const float4*)g0)[0];
    bReg[1] = ((const float4*)g0)[1];
    bReg[2] = ((const float4*)(g0 + N))[0];
    bReg[3] = ((const float4*)(g0 + N))[1];
  };

  auto store_stage = [&]() {
    bf16* sa = As + arow * 32 + acol;
    if constexpr (sizeof(TA) == 4) {
      float af[16] = {aRegF[0].x, aRegF[0].y, aRegF[0].z, aRegF[0].w,
                      aRegF[1].x, aRegF[1].y, aRegF[1].z, aRegF[1].w,
                      aRegF[2].x, aRegF[2].y, aRegF[2].z, aRegF[2].w,
                      aRegF[3].x, aRegF[3].y, aRegF[3].z, aRegF[3].w};
      v8bf p0, p1;
      #pragma unroll
      for (int i = 0; i < 8; ++i) { p0[i] = (bf16)af[i]; p1[i] = (bf16)af[8 + i]; }
      *(v8bf*)sa = p0;
      *(v8bf*)(sa + 8) = p1;
    } else {
      *(v8bf*)sa = aRegH[0];
      *(v8bf*)(sa + 8) = aRegH[1];
    }
    // B^T: pack (k, k+1) pair per n into one b32 store
    float lo[8] = {bReg[0].x, bReg[0].y, bReg[0].z, bReg[0].w,
                   bReg[1].x, bReg[1].y, bReg[1].z, bReg[1].w};
    float hi[8] = {bReg[2].x, bReg[2].y, bReg[2].z, bReg[2].w,
                   bReg[3].x, bReg[3].y, bReg[3].z, bReg[3].w};
    #pragma unroll
    for (int j = 0; j < 8; ++j) {
      v2bf p;
      p[0] = (bf16)lo[j];
      p[1] = (bf16)hi[j];
      *(v2bf*)(Bs + (size_t)(bnn + j) * 32 + bkk) = p;
    }
  };

  v8f acc[2][4];
  v8f zero = {};
  #pragma unroll
  for (int mi = 0; mi < 2; ++mi)
    #pragma unroll
    for (int ni = 0; ni < 4; ++ni) acc[mi][ni] = zero;

  load_stage(0);
  for (int k0 = 0; k0 < K; k0 += 32) {
    store_stage();
    __syncthreads();
    if (k0 + 32 < K) load_stage(k0 + 32);  // overlap with WMMA below

    v16bf afrag[2], bfrag[4];
    #pragma unroll
    for (int mi = 0; mi < 2; ++mi)
      afrag[mi] = load_a_frag(As + (wm + mi * 16) * 32, 32);
    #pragma unroll
    for (int ni = 0; ni < 4; ++ni)
      bfrag[ni] = load_b_frag(Bs + (wn + ni * 16) * 32, 32);

    #pragma unroll
    for (int mi = 0; mi < 2; ++mi)
      #pragma unroll
      for (int ni = 0; ni < 4; ++ni)
        acc[mi][ni] = wmma_bf16(afrag[mi], bfrag[ni], acc[mi][ni]);

    __syncthreads();
  }

  // Epilogue: C layout -> lane holds col = lane&15, rows r + 8*(lane>>4)
  const int lane = tid & 31;
  const int colL = lane & 15;
  const int rg = (lane >> 4) << 3;
  #pragma unroll
  for (int mi = 0; mi < 2; ++mi) {
    #pragma unroll
    for (int ni = 0; ni < 4; ++ni) {
      int gn = bn + wn + ni * 16 + colL;
      float bb = bias[gn];
      #pragma unroll
      for (int r = 0; r < 8; ++r) {
        int gm = bm + wm + mi * 16 + rg + r;
        C[(size_t)gm * N + gn] = acc[mi][ni][r] + bb;
      }
    }
  }
}

// ---------------------------------------------------------------------------
// RoPE + reorg. One wave per (b, s, head). Q scaled by 1/sqrt(D).
// Q  -> [B, NH,  S, D]  bf16 (A-operand layout source for Q*K^T)
// K  -> [B, NKV, S, D]  bf16 (natural layout == B^T layout for Q*K^T)
// V  -> [B, NKV, D, S]  bf16 (transposed  == B^T layout for P*V)
// ---------------------------------------------------------------------------
__global__ __launch_bounds__(256)
void rope_reorg_kernel(const float* __restrict__ qkv, bf16* __restrict__ q,
                       bf16* __restrict__ k, bf16* __restrict__ vt) {
  const int gw   = (blockIdx.x * blockDim.x + threadIdx.x) >> 5;
  const int lane = threadIdx.x & 31;
  const int head = gw % NHT;
  const int s    = (gw / NHT) % SEQ;
  const int b    = gw / (NHT * SEQ);
  if (b >= BATCH) return;

  const float* src = qkv + (size_t)(b * SEQ + s) * QKVN + head * HD;

  float out0, out1;
  if (head < NH + NKV) {
    // rotary part covers d in [0, 32): pairs (i, i+16)
    int i = lane & 15;
    float ang = (float)s * __expf(-(float)i * LOG_BASE_OVER_HALF_ROT);
    float sn, cs;
    __sincosf(ang, &sn, &cs);
    float x1 = src[i];
    float x2 = src[i + 16];
    out0 = (lane < 16) ? (x1 * cs - x2 * sn) : (x1 * sn + x2 * cs);
    out1 = src[lane + 32];  // pass-through half
  } else {
    out0 = src[lane];
    out1 = src[lane + 32];
  }

  if (head < NH) {
    bf16* dst = q + ((size_t)(b * NH + head) * SEQ + s) * HD;
    dst[lane]      = (bf16)(out0 * 0.125f);  // fold 1/sqrt(64) into Q
    dst[lane + 32] = (bf16)(out1 * 0.125f);
  } else if (head < NH + NKV) {
    bf16* dst = k + ((size_t)(b * NKV + (head - NH)) * SEQ + s) * HD;
    dst[lane]      = (bf16)out0;
    dst[lane + 32] = (bf16)out1;
  } else {
    int hv = head - NH - NKV;
    bf16* dst = vt + ((size_t)(b * NKV + hv) * HD) * SEQ + s;
    dst[(size_t)lane * SEQ]        = (bf16)out0;
    dst[(size_t)(lane + 32) * SEQ] = (bf16)out1;
  }
}

// ---------------------------------------------------------------------------
// Causal flash attention (GQA 4:1). Block = 4 waves, wave owns 16 q rows.
// Per 32-wide KV chunk: 4 WMMA (scores) + online softmax + 4 WMMA (P*V).
// Prefetches next chunk's K/V lines (global_prefetch_b8).
// Writes bf16 so the output projection can stage without conversion.
// ---------------------------------------------------------------------------
__global__ __launch_bounds__(128)
void flash_attn_kernel(const bf16* __restrict__ q, const bf16* __restrict__ kmat,
                       const bf16* __restrict__ vt, bf16* __restrict__ out) {
  __shared__ bf16 Ptile[4][16 * 40];  // per-wave P staging, ld 40 (16B aligned rows)

  const int wave = threadIdx.x >> 5;
  const int lane = threadIdx.x & 31;
  const int colL = lane & 15;
  const int rg   = (lane >> 4) << 3;  // 0 or 8
  const int qt   = blockIdx.x * 64 + wave * 16;
  const int h    = blockIdx.y;
  const int b    = blockIdx.z;
  const int hkv  = h >> 2;  // NH/NKV = 4

  const bf16* qbase = q + ((size_t)(b * NH + h) * SEQ + qt) * HD;
  v16bf qf0 = load_a_frag(qbase, HD);        // d = 0..31
  v16bf qf1 = load_a_frag(qbase + 32, HD);   // d = 32..63

  const bf16* kbase = kmat + ((size_t)(b * NKV + hkv) * SEQ) * HD;
  const bf16* vbase = vt + ((size_t)(b * NKV + hkv) * HD) * SEQ;

  float m[8], l[8];
  v8f oacc[4];
  v8f zero = {};
  #pragma unroll
  for (int r = 0; r < 8; ++r) { m[r] = -3.0e38f; l[r] = 0.0f; }
  #pragma unroll
  for (int ni = 0; ni < 4; ++ni) oacc[ni] = zero;

  bf16* P = &Ptile[wave][0];
  const int kv_end = qt + 16;  // causal bound (multiple of 16)

  for (int kv = 0; kv < kv_end; kv += 32) {
    // Prefetch next chunk's K rows and V rows into cache hierarchy
    if (kv + 32 < kv_end) {
      __builtin_prefetch(kbase + (size_t)(kv + 32) * HD, 0, 1);
      __builtin_prefetch(kbase + (size_t)(kv + 48) * HD, 0, 1);
      __builtin_prefetch(vbase + (kv + 32), 0, 1);
    }

    // ---- scores: two 16x16 tiles ----
    v8f sc[2];
    #pragma unroll
    for (int t = 0; t < 2; ++t) {
      int kvt = kv + t * 16;
      v8f c = zero;
      if (kvt < kv_end) {
        const bf16* kb = kbase + (size_t)kvt * HD;
        v16bf b0 = load_b_frag(kb, HD);       // d 0..31
        v16bf b1 = load_b_frag(kb + 32, HD);  // d 32..63
        c = wmma_bf16(qf0, b0, c);
        c = wmma_bf16(qf1, b1, c);
        int col = kvt + colL;
        #pragma unroll
        for (int r = 0; r < 8; ++r) {
          int row = qt + rg + r;
          if (col > row) c[r] = -1.0e4f;
        }
      } else {
        #pragma unroll
        for (int r = 0; r < 8; ++r) c[r] = -1.0e4f;
      }
      sc[t] = c;
    }

    // ---- online softmax (row spread across 16 lanes of each half-wave) ----
    #pragma unroll
    for (int r = 0; r < 8; ++r) {
      float mx = fmaxf(sc[0][r], sc[1][r]);
      #pragma unroll
      for (int off = 1; off < 16; off <<= 1) mx = fmaxf(mx, __shfl_xor(mx, off, 16));
      float mn = fmaxf(m[r], mx);
      float alpha = __expf(m[r] - mn);
      float p0 = __expf(sc[0][r] - mn);
      float p1 = __expf(sc[1][r] - mn);
      sc[0][r] = p0;
      sc[1][r] = p1;
      float rs = p0 + p1;
      #pragma unroll
      for (int off = 1; off < 16; off <<= 1) rs += __shfl_xor(rs, off, 16);
      l[r] = l[r] * alpha + rs;
      m[r] = mn;
      #pragma unroll
      for (int ni = 0; ni < 4; ++ni) oacc[ni][r] *= alpha;
    }

    // ---- P: C-layout -> LDS row-major -> A fragment ----
    #pragma unroll
    for (int t = 0; t < 2; ++t)
      #pragma unroll
      for (int r = 0; r < 8; ++r)
        P[(rg + r) * 40 + t * 16 + colL] = (bf16)sc[t][r];
    asm volatile("s_wait_dscnt 0" ::: "memory");
    v16bf pf = load_a_frag(P, 40);

    // ---- out += P * V ----
    #pragma unroll
    for (int ni = 0; ni < 4; ++ni) {
      v16bf vf = load_b_frag(vbase + (size_t)(ni * 16) * SEQ + kv, SEQ);
      oacc[ni] = wmma_bf16(pf, vf, oacc[ni]);
    }
  }

  // ---- epilogue: normalize and write [B, S, NH*HD] bf16 ----
  #pragma unroll
  for (int r = 0; r < 8; ++r) {
    int row = qt + rg + r;
    float inv = 1.0f / l[r];
    #pragma unroll
    for (int ni = 0; ni < 4; ++ni) {
      out[(size_t)(b * SEQ + row) * (NH * HD) + h * HD + ni * 16 + colL] =
          (bf16)(oacc[ni][r] * inv);
    }
  }
}

// ---------------------------------------------------------------------------
extern "C" void kernel_launch(void* const* d_in, const int* in_sizes, int n_in,
                              void* d_out, int out_size, void* d_ws, size_t ws_size,
                              hipStream_t stream) {
  (void)in_sizes; (void)n_in; (void)out_size; (void)ws_size;
  const float* x     = (const float*)d_in[0];
  const float* w_qkv = (const float*)d_in[1];
  const float* b_qkv = (const float*)d_in[2];
  const float* w_out = (const float*)d_in[3];
  const float* b_out = (const float*)d_in[4];
  float* out = (float*)d_out;

  const int M = BATCH * SEQ;  // 4096

  char* ws = (char*)d_ws;
  size_t o = 0;
  float* qkv = (float*)(ws + o); o += (size_t)M * QKVN * sizeof(float);              // 50.3 MB
  bf16*  qb  = (bf16*)(ws + o);  o += (size_t)BATCH * NH  * SEQ * HD * sizeof(bf16); // 16.8 MB
  bf16*  kb  = (bf16*)(ws + o);  o += (size_t)BATCH * NKV * SEQ * HD * sizeof(bf16); //  4.2 MB
  bf16*  vtb = (bf16*)(ws + o);  o += (size_t)BATCH * NKV * HD * SEQ * sizeof(bf16); //  4.2 MB
  bf16*  attn = (bf16*)(ws + o);                                                     // 16.8 MB

  // 1) QKV projection: [4096,2048](f32) x [2048,3072](f32) + bias
  gemm_bias_kernel<float><<<dim3(M / 128, QKVN / 128), 256, 0, stream>>>(
      x, w_qkv, b_qkv, qkv, M, QKVN, EMB);

  // 2) RoPE + layout reorg (one wave per (b,s,head))
  {
    int total_waves = BATCH * SEQ * NHT;              // 196608
    int blocks = (total_waves * 32) / 256;            // 24576
    rope_reorg_kernel<<<blocks, 256, 0, stream>>>(qkv, qb, kb, vtb);
  }

  // 3) Causal flash attention, GQA
  flash_attn_kernel<<<dim3(SEQ / 64, NH, BATCH), 128, 0, stream>>>(
      qb, kb, vtb, attn);

  // 4) Output projection: [4096,2048](bf16) x [2048,2048](f32) + bias -> d_out
  gemm_bias_kernel<bf16><<<dim3(M / 128, EMB / 128), 256, 0, stream>>>(
      attn, w_out, b_out, out, M, EMB, NH * HD);
}
